// LinearAdaMoleLayer_4999341932684
// MI455X (gfx1250) — matrix operand order repaired and checked by reading, not compiled, not run on hardware
//
#include <hip/hip_runtime.h>
#include <hip/hip_fp16.h>

typedef _Float16 v16h __attribute__((ext_vector_type(16)));
typedef _Float16 v8h  __attribute__((ext_vector_type(8)));
typedef _Float16 v4h  __attribute__((ext_vector_type(4)));
typedef float    v8f  __attribute__((ext_vector_type(8)));

constexpr int D_    = 4096;   // input dim
constexpr int O_    = 4096;   // output dim
constexpr int E_    = 8;      // experts
constexpr int R_    = 16;     // lora rank
constexpr int NTOK  = 8192;   // 4 * 2048 tokens
constexpr int KLORA = E_ * R_;           // 128
constexpr float MAX_THR = 0.125f;        // 1/8
constexpr float SCALING = 2.0f;          // 32/16

constexpr int BM = 128, BN = 128, BK = 32;
constexpr int LDK = 40;                    // halves per LDS row (16B aligned rows)
constexpr int NCHUNK = (D_ + KLORA) / BK;  // 132

// ---------------------------------------------------------------------------
// Kernel 1: gating (softmax gate, sigmoid threshold) + LoRA down-projection.
// Writes hw[token][e*R+r] = weight[token][e] * (x . A[e][r]) * SCALING
// ---------------------------------------------------------------------------
__global__ __launch_bounds__(256) void gate_lora_kernel(
    const float* __restrict__ x, const float* __restrict__ Wg,
    const float* __restrict__ Wt, const float* __restrict__ bt,
    const float* __restrict__ A, float* __restrict__ hw)
{
    constexpr int T = 2;                       // tokens per block
    __shared__ float xs[T][D_];                // 32 KB
    __shared__ float gsh[T][9];                // 8 gate logits + 1 thr logit
    __shared__ float wsh[T][E_];

    const int tid  = threadIdx.x;
    const int lane = tid & 31;
    const int wave = tid >> 5;                 // 0..7 (wave32)
    const size_t tok0 = (size_t)blockIdx.x * T;

    // stage T tokens of x into LDS
    for (int t = 0; t < T; ++t) {
        const float4* src = (const float4*)(x + (tok0 + t) * D_);
        float4* dst = (float4*)xs[t];
        for (int i = tid; i < D_ / 4; i += 256) dst[i] = src[i];
    }
    __syncthreads();

    // gating dot products: jobs 0..7 -> gates, job 8 -> threshold
    for (int job = wave; job < 9; job += 8) {
        const float* wrow = (job < 8) ? (Wg + (size_t)job * D_) : Wt;
        float s[T] = {};
        for (int k = lane; k < D_; k += 32) {
            float wv = wrow[k];
            #pragma unroll
            for (int t = 0; t < T; ++t) s[t] += xs[t][k] * wv;
        }
        #pragma unroll
        for (int t = 0; t < T; ++t) {
            float v = s[t];
            for (int off = 16; off > 0; off >>= 1) v += __shfl_xor(v, off, 32);
            if (lane == 0) gsh[t][job] = v;
        }
    }
    __syncthreads();

    // per-token adaptive mixture weights
    if (tid < T * E_) {
        const int t = tid >> 3, e = tid & 7;
        float mx = gsh[t][0];
        #pragma unroll
        for (int j = 1; j < 8; ++j) mx = fmaxf(mx, gsh[t][j]);
        float p[8]; float den = 0.f;
        #pragma unroll
        for (int j = 0; j < 8; ++j) { p[j] = expf(gsh[t][j] - mx); den += p[j]; }
        const float thr = MAX_THR / (1.f + expf(-(gsh[t][8] + bt[0])));
        float wsum = 0.f;
        #pragma unroll
        for (int j = 0; j < 8; ++j) {
            float aj = p[j] / den - thr;
            wsum += (aj >= 0.f) ? aj : 0.f;
        }
        if (wsum == 0.f) wsum = 1.f;
        float ae = p[e] / den - thr;
        wsh[t][e] = ((ae >= 0.f) ? ae : 0.f) / wsum;
    }
    __syncthreads();

    // LoRA down-projection: wave w handles expert e = w (16 rows of A)
    for (int r = 0; r < R_; ++r) {
        const float* Ar = A + ((size_t)wave * R_ + r) * D_;
        float s[T] = {};
        for (int k = lane; k < D_; k += 32) {
            float a = Ar[k];
            #pragma unroll
            for (int t = 0; t < T; ++t) s[t] += xs[t][k] * a;
        }
        #pragma unroll
        for (int t = 0; t < T; ++t) {
            float v = s[t];
            for (int off = 16; off > 0; off >>= 1) v += __shfl_xor(v, off, 32);
            if (lane == 0)
                hw[(tok0 + t) * KLORA + wave * R_ + r] = v * wsh[t][wave] * SCALING;
        }
    }
}

// ---------------------------------------------------------------------------
// Kernel 2: fused GEMM  out = x @ W_base^T + hw @ Bflat^T + b_base
// K = 4096 (base) + 128 (lora).  f32 emulated via 3x f16 WMMA split.
// Double-buffered LDS pipeline: fetch chunk kt+1 while computing chunk kt.
// ---------------------------------------------------------------------------
__device__ inline v8f wmma_f16(v16h a, v16h b, v8f c) {
    return __builtin_amdgcn_wmma_f32_16x16x32_f16(
        false, a, false, b, (short)0, c, false, false);
}

__device__ inline v16h cat8(v8h a, v8h b) {
    return __builtin_shufflevector(a, b, 0, 1, 2, 3, 4, 5, 6, 7,
                                         8, 9, 10, 11, 12, 13, 14, 15);
}

__device__ inline void split4(const float4 v, v4h& h, v4h& l) {
    const _Float16 h0 = (_Float16)v.x, h1 = (_Float16)v.y;
    const _Float16 h2 = (_Float16)v.z, h3 = (_Float16)v.w;
    h = (v4h){h0, h1, h2, h3};
    l = (v4h){(_Float16)(v.x - (float)h0), (_Float16)(v.y - (float)h1),
              (_Float16)(v.z - (float)h2), (_Float16)(v.w - (float)h3)};
}

// fetch A/B tiles for K-chunk starting at k0 into registers (4 float4 each)
__device__ inline void fetch_tiles(
    int k0, int rowg0, int colg0, int lrow, int lcol,
    const float* __restrict__ x,  const float* __restrict__ Wb,
    const float* __restrict__ Bm, const float* __restrict__ hw,
    float4* va, float4* vb)
{
    #pragma unroll
    for (int p = 0; p < 4; ++p) {
        const int r = lrow + p * 32;
        if (k0 < D_) {
            va[p] = *(const float4*)(x  + (size_t)(rowg0 + r) * D_ + k0 + lcol);
            vb[p] = *(const float4*)(Wb + (size_t)(colg0 + r) * D_ + k0 + lcol);
        } else {
            const int j = (k0 - D_) + lcol;         // 0..124
            va[p] = *(const float4*)(hw + (size_t)(rowg0 + r) * KLORA + j);
            const int e = j >> 4, rr = j & 15;      // Bflat[o][e*16+r] = B[e][o][r]
            vb[p] = *(const float4*)(Bm + ((size_t)e * O_ + (colg0 + r)) * R_ + rr);
        }
    }
}

// split registers into f16 hi/lo and store to one LDS buffer set
__device__ inline void store_tiles(
    _Float16* __restrict__ ah, _Float16* __restrict__ al,
    _Float16* __restrict__ bh, _Float16* __restrict__ bl,
    int lrow, int lcol, const float4* va, const float4* vb)
{
    #pragma unroll
    for (int p = 0; p < 4; ++p) {
        const int r = lrow + p * 32;
        v4h h, l;
        split4(va[p], h, l);
        *(v4h*)&ah[r * LDK + lcol] = h;
        *(v4h*)&al[r * LDK + lcol] = l;
        split4(vb[p], h, l);
        *(v4h*)&bh[r * LDK + lcol] = h;
        *(v4h*)&bl[r * LDK + lcol] = l;
    }
}

__global__ __launch_bounds__(256) void fused_gemm_kernel(
    const float* __restrict__ x,  const float* __restrict__ Wb,
    const float* __restrict__ bb, const float* __restrict__ Bm,
    const float* __restrict__ hw, float* __restrict__ out)
{
    // double-buffered LDS: 2 * 4 * 128*40*2B = 80 KB (of 320 KB/WGP)
    __shared__ __align__(16) _Float16 Ah[2][BM * LDK];
    __shared__ __align__(16) _Float16 Al[2][BM * LDK];
    __shared__ __align__(16) _Float16 Bh[2][BN * LDK];
    __shared__ __align__(16) _Float16 Bl[2][BN * LDK];

    const int tid  = threadIdx.x;
    const int lane = tid & 31;
    const int wave = tid >> 5;
    const int wm   = wave & 1;      // wave row   (2 x 64)
    const int wn   = wave >> 1;     // wave col   (4 x 32)
    const int sel  = lane >> 4;     // half-wave select
    const int ln   = lane & 15;

    const int rowg0 = blockIdx.y * BM;   // token rows
    const int colg0 = blockIdx.x * BN;   // output cols

    const int lrow = tid >> 3;           // 0..31
    const int lcol = (tid & 7) * 4;      // 0,4,...,28

    v8f acc[4][2];
    #pragma unroll
    for (int i = 0; i < 4; ++i)
        #pragma unroll
        for (int j = 0; j < 2; ++j)
            acc[i][j] = (v8f){0.f, 0.f, 0.f, 0.f, 0.f, 0.f, 0.f, 0.f};

    // prologue: stage chunk 0 into buffer 0
    float4 va[4], vb[4];
    fetch_tiles(0, rowg0, colg0, lrow, lcol, x, Wb, Bm, hw, va, vb);
    store_tiles(Ah[0], Al[0], Bh[0], Bl[0], lrow, lcol, va, vb);
    __syncthreads();

    #pragma unroll 1
    for (int kt = 0; kt < NCHUNK; ++kt) {
        const int  cur  = kt & 1;
        const bool more = (kt + 1) < NCHUNK;

        // issue next chunk's global loads first: latency hides under WMMAs
        if (more)
            fetch_tiles((kt + 1) * BK, rowg0, colg0, lrow, lcol,
                        x, Wb, Bm, hw, va, vb);

        // ---- fragments (contiguous k = sel*16 + i in A and B) + 24 WMMAs ------------
        v16h bfh[2], bfl[2];
        #pragma unroll
        for (int nt = 0; nt < 2; ++nt) {
            const int n = wn * 32 + nt * 16 + ln;
            const _Float16* pbh = &Bh[cur][n * LDK + sel * 16];  // B: k = sel*16 + i
            const _Float16* pbl = &Bl[cur][n * LDK + sel * 16];
            bfh[nt] = cat8(*(const v8h*)pbh, *(const v8h*)(pbh + 8));
            bfl[nt] = cat8(*(const v8h*)pbl, *(const v8h*)(pbl + 8));
        }
        #pragma unroll
        for (int mt = 0; mt < 4; ++mt) {
            const int m = wm * 64 + mt * 16 + ln;
            const _Float16* pah = &Ah[cur][m * LDK];   // A: k = sel*16 + i
            const _Float16* pal = &Al[cur][m * LDK];
            const v16h afh = cat8(*(const v8h*)(pah + sel * 16),
                                  *(const v8h*)(pah + sel * 16 + 8));
            const v16h afl = cat8(*(const v8h*)(pal + sel * 16),
                                  *(const v8h*)(pal + sel * 16 + 8));
            #pragma unroll
            for (int nt = 0; nt < 2; ++nt) {
                acc[mt][nt] = wmma_f16(afh, bfh[nt], acc[mt][nt]);  // hi*hi
                acc[mt][nt] = wmma_f16(afh, bfl[nt], acc[mt][nt]);  // hi*lo
                acc[mt][nt] = wmma_f16(afl, bfh[nt], acc[mt][nt]);  // lo*hi
            }
        }

        // stage next chunk into the other buffer; one barrier per iteration.
        // Safe: previous iteration's end barrier proved every wave consumed
        // buf[cur^1] fragments already.
        if (more) {
            const int nxt = cur ^ 1;
            store_tiles(Ah[nxt], Al[nxt], Bh[nxt], Bl[nxt], lrow, lcol, va, vb);
        }
        __syncthreads();
    }

    // ---- epilogue: + bias, store (C/D layout: row = j + 8*sel, col = ln) ----
    #pragma unroll
    for (int mt = 0; mt < 4; ++mt) {
        #pragma unroll
        for (int nt = 0; nt < 2; ++nt) {
            const int col = colg0 + wn * 32 + nt * 16 + ln;
            const float bias = bb[col];
            #pragma unroll
            for (int j = 0; j < 8; ++j) {
                const int row = rowg0 + wm * 64 + mt * 16 + sel * 8 + j;
                out[(size_t)row * O_ + col] = acc[mt][nt][j] + bias;
            }
        }
    }
}

// ---------------------------------------------------------------------------
extern "C" void kernel_launch(void* const* d_in, const int* in_sizes, int n_in,
                              void* d_out, int out_size, void* d_ws, size_t ws_size,
                              hipStream_t stream) {
    const float* x  = (const float*)d_in[0];   // [4,2048,4096]
    const float* Wb = (const float*)d_in[1];   // [4096,4096]
    const float* bb = (const float*)d_in[2];   // [4096]
    const float* Wg = (const float*)d_in[3];   // [8,4096]
    const float* Wt = (const float*)d_in[4];   // [1,4096]
    const float* bt = (const float*)d_in[5];   // [1]
    const float* A  = (const float*)d_in[6];   // [8,16,4096]
    const float* Bm = (const float*)d_in[7];   // [8,4096,16]
    float* out = (float*)d_out;                // [4,2048,4096]
    float* hw  = (float*)d_ws;                 // [8192,128] scratch (4 MB)

    gate_lora_kernel<<<NTOK / 2, 256, 0, stream>>>(x, Wg, Wt, bt, A, hw);

    dim3 grid(O_ / 128, NTOK / 128);           // (32, 64)
    fused_gemm_kernel<<<grid, 256, 0, stream>>>(x, Wb, bb, Bm, hw, out);
}
